// SHMarginLoss_57878979281484
// MI455X (gfx1250) — compile-verified, matrix operand order checked by reading, not run on hardware
//
#include <hip/hip_runtime.h>
#include <hip/hip_bf16.h>
#include <stdint.h>

// ---------------------------------------------------------------------------
// Semi-hard margin loss, fused: logits = x @ y^T never hits memory.
//   x, y : [N=4096][K=1024] fp32 ; out : scalar fp32
// Fast path: one-time fp32 -> bf16 hi/lo split into d_ws, then a
// double-buffered WMMA GEMM fed by GLOBAL_LOAD_ASYNC_TO_LDS_B128 (ASYNCcnt)
// with 3x v_wmma_f32_16x16x32_bf16 per K-chunk (hi*hi + hi*lo + lo*hi).
// Fallback (small ws): inline-conversion GEMM.
// ---------------------------------------------------------------------------

typedef __attribute__((ext_vector_type(16))) __bf16 v16bf;
typedef __attribute__((ext_vector_type(8)))  float  v8f;

#define KDIM   1024
#define MARGIN 1.0f

#if defined(__has_builtin)
#if __has_builtin(__builtin_amdgcn_global_load_async_to_lds_b128)
#define HAVE_ASYNC_LDS 1
#endif
#if __has_builtin(__builtin_amdgcn_s_wait_asynccnt)
#define HAVE_WAIT_ASYNC_BUILTIN 1
#endif
#endif

#ifdef HAVE_ASYNC_LDS
typedef int v4i __attribute__((vector_size(16)));
typedef __attribute__((address_space(1))) v4i v4i_as1;   // global
typedef __attribute__((address_space(3))) v4i v4i_as3;   // LDS
static __device__ __forceinline__ void async_cp16(const void* g, void* l) {
    __builtin_amdgcn_global_load_async_to_lds_b128(
        (v4i_as1*)(uintptr_t)g,
        (v4i_as3*)(unsigned)(uintptr_t)l, 0, 0);
}
static __device__ __forceinline__ void wait_async0() {
#ifdef HAVE_WAIT_ASYNC_BUILTIN
    __builtin_amdgcn_s_wait_asynccnt(0);
#else
    asm volatile("s_wait_asynccnt 0x0" ::: "memory");
#endif
}
#endif

// ---- order-preserving float <-> uint encoding (0 == "empty" sentinel) -----
__device__ __forceinline__ unsigned encf(float f) {
    unsigned u = __float_as_uint(f);
    return (u & 0x80000000u) ? ~u : (u | 0x80000000u);
}
__device__ __forceinline__ float decf(unsigned e) {
    unsigned u = (e & 0x80000000u) ? (e & 0x7FFFFFFFu) : ~e;
    return __uint_as_float(u);
}

// ---- deterministic off-diagonal index (stand-in for jax.random) -----------
__device__ __forceinline__ unsigned hash_u32(unsigned s) {
    s ^= s >> 16; s *= 0x7feb352du;
    s ^= s >> 15; s *= 0x846ca68bu;
    s ^= s >> 16; return s;
}
__device__ __forceinline__ int offdiag(int k, unsigned salt, int n) {
    int j = (int)(hash_u32((unsigned)k ^ salt) % (unsigned)(n - 1));
    return j + (j >= k);
}

__device__ __forceinline__ v8f wmma_bf16(v16bf a, v16bf b, v8f c) {
    return __builtin_amdgcn_wmma_f32_16x16x32_bf16(
        false, a, false, b, (short)0, c, false, false);
}

union FB { v16bf v; uint4 u[2]; };
union BP4 { __bf16 b[4]; uint2 u; };

// ===========================================================================
// Kernel 0: zero the encoded row/col max arrays (2N u32)
// ===========================================================================
__global__ void k_init(unsigned* enc, int n2) {
    int i = blockIdx.x * blockDim.x + threadIdx.x;
    if (i < n2) enc[i] = 0u;
}

// ===========================================================================
// Kernel 0b: fp32 -> bf16 hi/lo split (one float4 per thread)
// ===========================================================================
__global__ void k_convert(const float* __restrict__ in,
                          __bf16* __restrict__ hi, __bf16* __restrict__ lo,
                          int n4) {
    int i = blockIdx.x * blockDim.x + threadIdx.x;
    if (i >= n4) return;
    float4 v = ((const float4*)in)[i];
    BP4 H, L;
    float f0 = v.x, f1 = v.y, f2 = v.z, f3 = v.w;
    __bf16 h0 = (__bf16)f0, h1 = (__bf16)f1, h2 = (__bf16)f2, h3 = (__bf16)f3;
    H.b[0] = h0; H.b[1] = h1; H.b[2] = h2; H.b[3] = h3;
    L.b[0] = (__bf16)(f0 - (float)h0);
    L.b[1] = (__bf16)(f1 - (float)h1);
    L.b[2] = (__bf16)(f2 - (float)h2);
    L.b[3] = (__bf16)(f3 - (float)h3);
    ((uint2*)hi)[i] = H.u;
    ((uint2*)lo)[i] = L.u;
}

// ===========================================================================
// Kernel 1: one wave per row k — exact fp32 dots for anchors + imposters
// ===========================================================================
__global__ void k_dots(const float* __restrict__ x, const float* __restrict__ y,
                       float* __restrict__ anchor,
                       float* __restrict__ imp1r, float* __restrict__ fallr,
                       float* __restrict__ imp1c, float* __restrict__ fallc,
                       int n, int kd) {
    int lane = threadIdx.x & 31;
    int k = blockIdx.x * (blockDim.x >> 5) + (threadIdx.x >> 5);
    if (k >= n) return;

    int j1 = offdiag(k, 0x9E3779B9u, n);
    int j2 = offdiag(k, 0x85EBCA6Bu, n);
    int j3 = offdiag(k, 0xC2B2AE35u, n);
    int j4 = offdiag(k, 0x27D4EB2Fu, n);

    const float* xk = x + (size_t)k * kd;
    const float* yk = y + (size_t)k * kd;
    const float* y1 = y + (size_t)j1 * kd;
    const float* y2 = y + (size_t)j2 * kd;
    const float* x3 = x + (size_t)j3 * kd;
    const float* x4 = x + (size_t)j4 * kd;

    float a = 0.f, r1 = 0.f, r2 = 0.f, c1 = 0.f, c2 = 0.f;
    for (int i = lane; i < kd; i += 32) {
        float xv = xk[i], yv = yk[i];
        a  = fmaf(xv, yv,    a);
        r1 = fmaf(xv, y1[i], r1);
        r2 = fmaf(xv, y2[i], r2);
        c1 = fmaf(x3[i], yv, c1);
        c2 = fmaf(x4[i], yv, c2);
    }
    for (int s = 16; s >= 1; s >>= 1) {
        a  += __shfl_down(a,  s, 32);
        r1 += __shfl_down(r1, s, 32);
        r2 += __shfl_down(r2, s, 32);
        c1 += __shfl_down(c1, s, 32);
        c2 += __shfl_down(c2, s, 32);
    }
    if (lane == 0) {
        anchor[k] = a;
        imp1r[k] = r1; fallr[k] = r2;
        imp1c[k] = c1; fallc[k] = c2;
    }
}

// ===========================================================================
// Shared epilogue: masked row/col max (semi-hard candidate)
// ===========================================================================
__device__ __forceinline__ void epilogue_reduce(
        v8f acc[2][2], int t, int lane, int wm, int wn, int M0, int N0,
        const float* sAncR, const float* sAncC,
        unsigned* sRowMax, unsigned* sColMax,
        unsigned* rowEnc, unsigned* colEnc) {
    const int rhalf = (lane >> 4) << 3;

    // column partials
    #pragma unroll
    for (int an = 0; an < 2; ++an) {
        int ci = wn * 32 + an * 16 + (lane & 15);
        float ac = sAncC[ci];
        int gn = N0 + ci;
        unsigned m = 0u;
        #pragma unroll
        for (int am = 0; am < 2; ++am)
            #pragma unroll
            for (int r = 0; r < 8; ++r) {
                int gm = M0 + wm * 32 + am * 16 + r + rhalf;
                float v = acc[am][an][r];
                if (gm != gn && v < ac) {
                    unsigned e = encf(v);
                    m = (e > m) ? e : m;
                }
            }
        atomicMax(&sColMax[ci], m);
    }

    // row partials: shuffle-reduce over 16 lanes of each half
    #pragma unroll
    for (int am = 0; am < 2; ++am)
        #pragma unroll
        for (int r = 0; r < 8; ++r) {
            int ri = wm * 32 + am * 16 + r + rhalf;
            float ar = sAncR[ri];
            int gm = M0 + ri;
            unsigned m = 0u;
            #pragma unroll
            for (int an = 0; an < 2; ++an) {
                int gn = N0 + wn * 32 + an * 16 + (lane & 15);
                float v = acc[am][an][r];
                if (gm != gn && v < ar) {
                    unsigned e = encf(v);
                    m = (e > m) ? e : m;
                }
            }
            #pragma unroll
            for (int s = 8; s >= 1; s >>= 1) {
                unsigned o = __shfl_xor(m, s, 32);
                m = (o > m) ? o : m;
            }
            if ((lane & 15) == 0) atomicMax(&sRowMax[ri], m);
        }

    __syncthreads();
    if (t < 128)      atomicMax(&rowEnc[M0 + t], sRowMax[t]);
    else if (t < 192) atomicMax(&colEnc[N0 + (t - 128)], sColMax[t - 128]);
}

// ===========================================================================
// Kernel 2 (fast): double-buffered WMMA GEMM on pre-split bf16, async
// global->LDS staging. Block tile 128x64, 8 waves (4x2), wave = 32x32.
// ===========================================================================
__global__ void __launch_bounds__(256)
k_gemm_bf16(const __bf16* __restrict__ xhi, const __bf16* __restrict__ xlo,
            const __bf16* __restrict__ yhi, const __bf16* __restrict__ ylo,
            const float* __restrict__ anchor,
            unsigned* __restrict__ rowEnc, unsigned* __restrict__ colEnc,
            int kd) {
    __shared__ __bf16 sAhi[2][128][40];
    __shared__ __bf16 sAlo[2][128][40];
    __shared__ __bf16 sBhi[2][64][40];
    __shared__ __bf16 sBlo[2][64][40];
    __shared__ float sAncR[128];
    __shared__ float sAncC[64];
    __shared__ unsigned sRowMax[128];
    __shared__ unsigned sColMax[64];

    const int t    = threadIdx.x;
    const int lane = t & 31;
    const int wid  = t >> 5;
    const int wm   = wid & 3;
    const int wn   = wid >> 2;
    const int M0   = blockIdx.y * 128;
    const int N0   = blockIdx.x * 64;

    if (t < 128) { sAncR[t] = anchor[M0 + t]; sRowMax[t] = 0u; }
    if (t < 64)  { sAncC[t] = anchor[N0 + t]; }
    if (t >= 128 && t < 192) sColMax[t - 128] = 0u;

    v8f acc[2][2];
    #pragma unroll
    for (int i = 0; i < 2; ++i)
        #pragma unroll
        for (int j = 0; j < 2; ++j)
            acc[i][j] = (v8f){0.f, 0.f, 0.f, 0.f, 0.f, 0.f, 0.f, 0.f};

    const int rsel = lane & 15;
    const int koff = (lane >> 4) << 3;

    // per-thread copy slots: A -> 2x (hi,lo), B -> 1x (hi,lo) : 16B each
    const int arow0 = t >> 2;                 // 0..63
    const int arow1 = (t + 256) >> 2;         // 64..127
    const int akq   = (t & 3) << 3;           // 0,8,16,24 (bf16 elems)
    const int brow  = t >> 2;                 // 0..63
    const int bkq   = akq;

    auto issue = [&](int buf, int kc) {
        const __bf16* gah0 = xhi + (size_t)(M0 + arow0) * kd + kc + akq;
        const __bf16* gah1 = xhi + (size_t)(M0 + arow1) * kd + kc + akq;
        const __bf16* gal0 = xlo + (size_t)(M0 + arow0) * kd + kc + akq;
        const __bf16* gal1 = xlo + (size_t)(M0 + arow1) * kd + kc + akq;
        const __bf16* gbh  = yhi + (size_t)(N0 + brow) * kd + kc + bkq;
        const __bf16* gbl  = ylo + (size_t)(N0 + brow) * kd + kc + bkq;
#ifdef HAVE_ASYNC_LDS
        async_cp16(gah0, &sAhi[buf][arow0][akq]);
        async_cp16(gah1, &sAhi[buf][arow1][akq]);
        async_cp16(gal0, &sAlo[buf][arow0][akq]);
        async_cp16(gal1, &sAlo[buf][arow1][akq]);
        async_cp16(gbh,  &sBhi[buf][brow][bkq]);
        async_cp16(gbl,  &sBlo[buf][brow][bkq]);
#else
        *(uint4*)&sAhi[buf][arow0][akq] = *(const uint4*)gah0;
        *(uint4*)&sAhi[buf][arow1][akq] = *(const uint4*)gah1;
        *(uint4*)&sAlo[buf][arow0][akq] = *(const uint4*)gal0;
        *(uint4*)&sAlo[buf][arow1][akq] = *(const uint4*)gal1;
        *(uint4*)&sBhi[buf][brow][bkq]  = *(const uint4*)gbh;
        *(uint4*)&sBlo[buf][brow][bkq]  = *(const uint4*)gbl;
#endif
    };

    const int nchunks = kd >> 5;
    int buf = 0;
    issue(0, 0);

    for (int ic = 0; ic < nchunks; ++ic) {
#ifdef HAVE_ASYNC_LDS
        wait_async0();             // this wave's async writes landed
#endif
        __syncthreads();           // all waves' staging visible; prev compute done
        if (ic + 1 < nchunks) issue(buf ^ 1, (ic + 1) << 5);

        v16bf aH[2], aL[2], bH[2], bL[2];
        #pragma unroll
        for (int am = 0; am < 2; ++am) {
            int row = wm * 32 + am * 16 + rsel;
            FB fh, fl;
            const __bf16* ph = &sAhi[buf][row][koff];
            const __bf16* pl = &sAlo[buf][row][koff];
            fh.u[0] = *(const uint4*)ph;  fh.u[1] = *(const uint4*)(ph + 16);
            fl.u[0] = *(const uint4*)pl;  fl.u[1] = *(const uint4*)(pl + 16);
            aH[am] = fh.v; aL[am] = fl.v;
        }
        #pragma unroll
        for (int an = 0; an < 2; ++an) {
            int row = wn * 32 + an * 16 + rsel;
            FB fh, fl;
            const __bf16* ph = &sBhi[buf][row][koff];
            const __bf16* pl = &sBlo[buf][row][koff];
            fh.u[0] = *(const uint4*)ph;  fh.u[1] = *(const uint4*)(ph + 16);
            fl.u[0] = *(const uint4*)pl;  fl.u[1] = *(const uint4*)(pl + 16);
            bH[an] = fh.v; bL[an] = fl.v;
        }

        #pragma unroll
        for (int am = 0; am < 2; ++am)
            #pragma unroll
            for (int an = 0; an < 2; ++an) {
                acc[am][an] = wmma_bf16(aH[am], bH[an], acc[am][an]);
                acc[am][an] = wmma_bf16(aH[am], bL[an], acc[am][an]);
                acc[am][an] = wmma_bf16(aL[am], bH[an], acc[am][an]);
            }
        buf ^= 1;
    }

    epilogue_reduce(acc, t, lane, wm, wn, M0, N0,
                    sAncR, sAncC, sRowMax, sColMax, rowEnc, colEnc);
}

// ===========================================================================
// Kernel 2 (fallback, small ws): inline-conversion GEMM
// ===========================================================================
__global__ void __launch_bounds__(256)
k_gemm_f32(const float* __restrict__ x, const float* __restrict__ y,
           const float* __restrict__ anchor,
           unsigned* __restrict__ rowEnc, unsigned* __restrict__ colEnc,
           int kd) {
    __shared__ __bf16 sAhi[128][40];
    __shared__ __bf16 sAlo[128][40];
    __shared__ __bf16 sBhi[64][40];
    __shared__ __bf16 sBlo[64][40];
    __shared__ float sAncR[128];
    __shared__ float sAncC[64];
    __shared__ unsigned sRowMax[128];
    __shared__ unsigned sColMax[64];

    const int t    = threadIdx.x;
    const int lane = t & 31;
    const int wid  = t >> 5;
    const int wm   = wid & 3;
    const int wn   = wid >> 2;
    const int M0   = blockIdx.y * 128;
    const int N0   = blockIdx.x * 64;

    if (t < 128) { sAncR[t] = anchor[M0 + t]; sRowMax[t] = 0u; }
    if (t < 64)  { sAncC[t] = anchor[N0 + t]; }
    if (t >= 128 && t < 192) sColMax[t - 128] = 0u;

    v8f acc[2][2];
    for (int i = 0; i < 2; ++i)
        for (int j = 0; j < 2; ++j)
            acc[i][j] = (v8f){0.f, 0.f, 0.f, 0.f, 0.f, 0.f, 0.f, 0.f};

    const int rsel = lane & 15;
    const int koff = (lane >> 4) << 3;

    for (int kc = 0; kc < kd; kc += 32) {
        __syncthreads();
        #pragma unroll
        for (int i = 0; i < 4; ++i) {
            int li  = t + (i << 8);
            int row = li >> 3;
            int kq  = (li & 7) << 2;
            float4 v = *(const float4*)(x + (size_t)(M0 + row) * kd + kc + kq);
            BP4 H, L;
            float f0 = v.x, f1 = v.y, f2 = v.z, f3 = v.w;
            __bf16 h0 = (__bf16)f0, h1 = (__bf16)f1, h2 = (__bf16)f2, h3 = (__bf16)f3;
            H.b[0] = h0; H.b[1] = h1; H.b[2] = h2; H.b[3] = h3;
            L.b[0] = (__bf16)(f0 - (float)h0);
            L.b[1] = (__bf16)(f1 - (float)h1);
            L.b[2] = (__bf16)(f2 - (float)h2);
            L.b[3] = (__bf16)(f3 - (float)h3);
            *(uint2*)&sAhi[row][kq] = H.u;
            *(uint2*)&sAlo[row][kq] = L.u;
        }
        #pragma unroll
        for (int i = 0; i < 2; ++i) {
            int li  = t + (i << 8);
            int row = li >> 3;
            int kq  = (li & 7) << 2;
            float4 v = *(const float4*)(y + (size_t)(N0 + row) * kd + kc + kq);
            BP4 H, L;
            float f0 = v.x, f1 = v.y, f2 = v.z, f3 = v.w;
            __bf16 h0 = (__bf16)f0, h1 = (__bf16)f1, h2 = (__bf16)f2, h3 = (__bf16)f3;
            H.b[0] = h0; H.b[1] = h1; H.b[2] = h2; H.b[3] = h3;
            L.b[0] = (__bf16)(f0 - (float)h0);
            L.b[1] = (__bf16)(f1 - (float)h1);
            L.b[2] = (__bf16)(f2 - (float)h2);
            L.b[3] = (__bf16)(f3 - (float)h3);
            *(uint2*)&sBhi[row][kq] = H.u;
            *(uint2*)&sBlo[row][kq] = L.u;
        }
        __syncthreads();

        v16bf aH[2], aL[2], bH[2], bL[2];
        #pragma unroll
        for (int am = 0; am < 2; ++am) {
            int row = wm * 32 + am * 16 + rsel;
            FB fh, fl;
            const __bf16* ph = &sAhi[row][koff];
            const __bf16* pl = &sAlo[row][koff];
            fh.u[0] = *(const uint4*)ph;  fh.u[1] = *(const uint4*)(ph + 16);
            fl.u[0] = *(const uint4*)pl;  fl.u[1] = *(const uint4*)(pl + 16);
            aH[am] = fh.v; aL[am] = fl.v;
        }
        #pragma unroll
        for (int an = 0; an < 2; ++an) {
            int row = wn * 32 + an * 16 + rsel;
            FB fh, fl;
            const __bf16* ph = &sBhi[row][koff];
            const __bf16* pl = &sBlo[row][koff];
            fh.u[0] = *(const uint4*)ph;  fh.u[1] = *(const uint4*)(ph + 16);
            fl.u[0] = *(const uint4*)pl;  fl.u[1] = *(const uint4*)(pl + 16);
            bH[an] = fh.v; bL[an] = fl.v;
        }

        #pragma unroll
        for (int am = 0; am < 2; ++am)
            #pragma unroll
            for (int an = 0; an < 2; ++an) {
                acc[am][an] = wmma_bf16(aH[am], bH[an], acc[am][an]);
                acc[am][an] = wmma_bf16(aH[am], bL[an], acc[am][an]);
                acc[am][an] = wmma_bf16(aL[am], bH[an], acc[am][an]);
            }
    }

    epilogue_reduce(acc, t, lane, wm, wn, M0, N0,
                    sAncR, sAncC, sRowMax, sColMax, rowEnc, colEnc);
}

// ===========================================================================
// Kernel 3: final hinge terms + mean (single block)
// ===========================================================================
__global__ void k_finalize(const float* __restrict__ anchor,
                           const float* __restrict__ imp1r, const float* __restrict__ fallr,
                           const float* __restrict__ imp1c, const float* __restrict__ fallc,
                           const unsigned* __restrict__ rowEnc,
                           const unsigned* __restrict__ colEnc,
                           float* __restrict__ out, int n) {
    __shared__ float red[256];
    int t = threadIdx.x;
    float sum = 0.f;
    for (int k = t; k < n; k += 256) {
        float a = anchor[k];
        float d1 = fmaxf(imp1r[k] - a + MARGIN, 0.f);
        unsigned e = rowEnc[k];
        float imp2 = e ? decf(e) : fallr[k];
        float d2 = fmaxf(imp2 - a + MARGIN, 0.f);
        sum += d1 + d2;
        float d1c = fmaxf(imp1c[k] - a + MARGIN, 0.f);
        unsigned ec = colEnc[k];
        float imp2c = ec ? decf(ec) : fallc[k];
        float d2c = fmaxf(imp2c - a + MARGIN, 0.f);
        sum += d1c + d2c;
    }
    red[t] = sum;
    __syncthreads();
    for (int s = 128; s >= 1; s >>= 1) {
        if (t < s) red[t] += red[t + s];
        __syncthreads();
    }
    if (t == 0) out[0] = red[0] / (float)n;
}

// ===========================================================================
extern "C" void kernel_launch(void* const* d_in, const int* in_sizes, int n_in,
                              void* d_out, int out_size, void* d_ws, size_t ws_size,
                              hipStream_t stream) {
    const float* x = (const float*)d_in[0];
    const float* y = (const float*)d_in[1];
    float* out = (float*)d_out;

    const int kd = KDIM;
    const int n  = in_sizes[0] / kd;   // 4096

    // workspace layout: 7n floats (scalars) then 4 bf16 matrices (fast path)
    float* ws      = (float*)d_ws;
    float* anchor  = ws;
    float* imp1r   = ws + (size_t)n;
    float* fallr   = ws + (size_t)2 * n;
    float* imp1c   = ws + (size_t)3 * n;
    float* fallc   = ws + (size_t)4 * n;
    unsigned* rowEnc = (unsigned*)(ws + (size_t)5 * n);
    unsigned* colEnc = rowEnc + n;

    const size_t elems     = (size_t)n * kd;
    const size_t needFast  = (size_t)7 * n * 4 + 4 * elems * 2;
    const bool   fast      = ws_size >= needFast;

    // 0) zero encoded max arrays
    k_init<<<(2 * n + 255) / 256, 256, 0, stream>>>(rowEnc, 2 * n);

    // 1) anchors + gathered imposters (one wave per row)
    k_dots<<<n / 8, 256, 0, stream>>>(x, y, anchor, imp1r, fallr, imp1c, fallc, n, kd);

    // 2) fused WMMA GEMM + masked row/col max
    dim3 grid(n / 64, n / 128);
    if (fast) {
        __bf16* bfb = (__bf16*)(ws + (size_t)7 * n);
        __bf16* xhi = bfb;
        __bf16* xlo = bfb + elems;
        __bf16* yhi = bfb + 2 * elems;
        __bf16* ylo = bfb + 3 * elems;
        int n4 = (int)(elems / 4);
        k_convert<<<(n4 + 255) / 256, 256, 0, stream>>>(x, xhi, xlo, n4);
        k_convert<<<(n4 + 255) / 256, 256, 0, stream>>>(y, yhi, ylo, n4);
        k_gemm_bf16<<<grid, 256, 0, stream>>>(xhi, xlo, yhi, ylo, anchor,
                                              rowEnc, colEnc, kd);
    } else {
        k_gemm_f32<<<grid, 256, 0, stream>>>(x, y, anchor, rowEnc, colEnc, kd);
    }

    // 3) finalize scalar loss
    k_finalize<<<1, 256, 0, stream>>>(anchor, imp1r, fallr, imp1c, fallc,
                                      rowEnc, colEnc, out, n);
}
